// TransformerLayer_9732395893255
// MI455X (gfx1250) — matrix operand order repaired and checked by reading, not compiled, not run on hardware
//
#include <hip/hip_runtime.h>
#include <hip/hip_bf16.h>

#define DEVINL __device__ __forceinline__

typedef __attribute__((ext_vector_type(16))) __bf16 v16bf;
typedef __attribute__((ext_vector_type(8)))  float  v8f;

union AFrag { v16bf v; unsigned short u[16]; };
union CFrag { v8f  v; float f[8]; };

constexpr int Bc  = 8;
constexpr int Sc  = 4096;
constexpr int Dc  = 512;
constexpr int Hc  = 8;
constexpr int DKc = 64;
constexpr int DFFc = 1024;
constexpr int KSc = 3;
constexpr int Mtot = Bc * Sc;   // 32768 token rows

DEVINL unsigned short f2bf(float f) {
  unsigned int u = __float_as_uint(f);
  u = u + 0x7FFFu + ((u >> 16) & 1u);         // round-to-nearest-even
  return (unsigned short)(u >> 16);
}
DEVINL float bf2f(unsigned short s) {
  return __uint_as_float(((unsigned int)s) << 16);
}
// A-fragment K index for 16-bit 16x32 A tile (ISA 7.12.2): lane half hh,
// element i -> K = i + 8*hh (i<8) or (i-8)+16+8*hh (i>=8)
DEVINL int amap(int i, int hh) { return i + 8 * hh + (i >= 8 ? 8 : 0); }

// CDNA5 async global->LDS copy (ASYNCcnt-tracked, GV addressing mode).
// lds = byte offset into LDS (low 32 bits of generic shared address),
// g   = 64-bit global address; copies 16 bytes per active lane.
DEVINL void async_b128(unsigned lds, const void* g) {
  asm volatile("global_load_async_to_lds_b128 %0, %1, off"
               :: "v"(lds), "v"((unsigned long long)(size_t)g)
               : "memory");
}
DEVINL void async_wait0() {
  asm volatile("s_wait_asynccnt 0x0" ::: "memory");
}

// ---------------------------------------------------------------------------
// fp32 -> bf16 conversion
// ---------------------------------------------------------------------------
__global__ __launch_bounds__(256) void cvt_kernel(const float* __restrict__ in,
                                                  unsigned short* __restrict__ out,
                                                  int n) {
  int i = blockIdx.x * 256 + threadIdx.x;
  if (i < n) out[i] = f2bf(in[i]);
}

// ---------------------------------------------------------------------------
// Tiled WMMA GEMM:  C[M,N] = A[M,K] * W[N,K]^T + bias   (torch Linear)
// 128x128 tile / workgroup, 256 threads = 8 waves, each wave owns 32x64.
// Double-buffered LDS fed by global_load_async_to_lds_b128; one barrier +
// one s_wait_asynccnt per K-step; WMMAs overlap the async tile fetch.
// mode 0: fp32 out.  mode 1: exact-GELU^2 epilogue, bf16 out.
// ---------------------------------------------------------------------------
__global__ __launch_bounds__(256) void gemm_bf16_wmma(
    const unsigned short* __restrict__ A,   // [M,K] bf16
    const unsigned short* __restrict__ W,   // [N,K] bf16
    const float* __restrict__ bias,         // [N]
    float* __restrict__ outF,
    unsigned short* __restrict__ outB,
    int K, int N, int mode) {
  __shared__ unsigned short As[2][128][40];  // 80B pitch: 16B aligned, skewed
  __shared__ unsigned short Ws[2][128][40];
  const int tid  = threadIdx.x;
  const int lane = tid & 31;
  const int wave = tid >> 5;                 // 0..7
  const int wm   = (wave >> 1) * 32;         // wave row offset in tile
  const int wn   = (wave & 1) * 64;          // wave col offset in tile
  const int hh   = lane >> 4;
  const int m16  = lane & 15;
  const int row0 = blockIdx.y * 128;
  const int col0 = blockIdx.x * 128;

  // per-thread async copy slot: row r, two 8-bf16 chunks at cs, cs+8
  const int r  = tid >> 1;                   // 0..127
  const int cs = (tid & 1) * 16;             // 0 or 16
  const unsigned asBase = (unsigned)(size_t)&As[0][0][0];
  const unsigned wsBase = (unsigned)(size_t)&Ws[0][0][0];
  constexpr unsigned BUFB = 128u * 40u * 2u; // bytes per LDS buffer

  CFrag c[2][4];
#pragma unroll
  for (int ar = 0; ar < 2; ++ar)
#pragma unroll
    for (int t = 0; t < 4; ++t)
#pragma unroll
      for (int q = 0; q < 8; ++q) c[ar][t].f[q] = 0.f;

  auto issue = [&](int k0, int buf) {
    const unsigned ab = asBase + (unsigned)buf * BUFB + (unsigned)(r * 80 + cs * 2);
    const unsigned wb = wsBase + (unsigned)buf * BUFB + (unsigned)(r * 80 + cs * 2);
    const unsigned short* ag = &A[(size_t)(row0 + r) * K + k0 + cs];
    const unsigned short* wg = &W[(size_t)(col0 + r) * K + k0 + cs];
    async_b128(ab,      ag);
    async_b128(ab + 16, ag + 8);
    async_b128(wb,      wg);
    async_b128(wb + 16, wg + 8);
  };

  issue(0, 0);
  // one-shot L2 prefetch of the second K tile (no loop duplication)
  __builtin_prefetch(&A[(size_t)(row0 + r) * K + 32 + cs], 0, 1);
  __builtin_prefetch(&W[(size_t)(col0 + r) * K + 32 + cs], 0, 1);

  int buf = 0;
  for (int k0 = 0; k0 < K; k0 += 32, buf ^= 1) {
    async_wait0();          // my tile-k copies have landed in LDS
    __syncthreads();        // everyone's copies landed; prior reads done
    if (k0 + 32 < K) issue(k0 + 32, buf ^ 1);   // prefetch next tile async

    AFrag a[2];
#pragma unroll
    for (int ar = 0; ar < 2; ++ar)
#pragma unroll
      for (int i = 0; i < 16; ++i)
        a[ar].u[i] = As[buf][wm + ar * 16 + m16][amap(i, hh)];
#pragma unroll
    for (int t = 0; t < 4; ++t) {
      AFrag bf;
#pragma unroll
      for (int i = 0; i < 16; ++i) bf.u[i] = Ws[buf][wn + t * 16 + m16][amap(i, hh)];
#pragma unroll
      for (int ar = 0; ar < 2; ++ar)
        c[ar][t].v = __builtin_amdgcn_wmma_f32_16x16x32_bf16(
            false, a[ar].v, false, bf.v, (short)0, c[ar][t].v, false, false);
    }
  }

#pragma unroll
  for (int t = 0; t < 4; ++t) {
    int n = col0 + wn + t * 16 + m16;
    float bb = bias[n];
#pragma unroll
    for (int ar = 0; ar < 2; ++ar)
#pragma unroll
      for (int q = 0; q < 8; ++q) {
        int m = row0 + wm + ar * 16 + q + 8 * hh;
        size_t o = (size_t)m * N + n;
        float v = c[ar][t].f[q] + bb;
        if (mode == 0) {
          outF[o] = v;
        } else {
          float g = 0.5f * v * (1.f + erff(v * 0.70710678118f));
          outB[o] = f2bf(g * g);
        }
      }
  }
}

// ---------------------------------------------------------------------------
// Causal depthwise conv over sequence (KS=3, left pad) on [B,S,H,DK] channels.
// act=1: elu(y)+1 feature map (q/k); act=0: plain (v). Output bf16.
// ---------------------------------------------------------------------------
__global__ __launch_bounds__(256) void dwconv_kernel(
    const float* __restrict__ raw, const float* __restrict__ w,
    const float* __restrict__ bias, unsigned short* __restrict__ outb, int act) {
  int idx = blockIdx.x * 256 + threadIdx.x;       // exactly B*S*D threads
  int cch = idx & (Dc - 1);
  int bs  = idx / Dc;
  int s   = bs & (Sc - 1);
  int dk  = cch & (DKc - 1);
  float y = bias[dk];
#pragma unroll
  for (int j = 0; j < KSc; ++j) {
    int sp = s - (KSc - 1) + j;
    if (sp >= 0)
      y += w[dk * KSc + j] * raw[(size_t)(bs - (KSc - 1) + j) * Dc + cch];
  }
  if (act) y = (y > 0.f) ? (y + 1.f) : expf(y);   // elu(y)+1
  outb[idx] = f2bf(y);
}

// ---------------------------------------------------------------------------
// ksum[bh][d] = sum_s K[b,s,h,d]
// ---------------------------------------------------------------------------
__global__ __launch_bounds__(256) void ksum_kernel(const unsigned short* __restrict__ kb,
                                                   float* __restrict__ ksum) {
  int bh = blockIdx.x;
  int b = bh / Hc, h = bh % Hc;
  int d = threadIdx.x & 63, seg = threadIdx.x >> 6;
  float acc = 0.f;
  for (int s = seg; s < Sc; s += 4)
    acc += bf2f(kb[(size_t)(b * Sc + s) * Dc + h * DKc + d]);
  __shared__ float red[256];
  red[threadIdx.x] = acc;
  __syncthreads();
  if (seg == 0)
    ksum[bh * DKc + d] = red[d] + red[64 + d] + red[128 + d] + red[192 + d];
}

// ---------------------------------------------------------------------------
// z[bh][s] = 1 / sum_d Q[s,d]*(ksum[d]+eps)
// ---------------------------------------------------------------------------
__global__ __launch_bounds__(256) void z_kernel(const unsigned short* __restrict__ qb,
                                                const float* __restrict__ ksum,
                                                float* __restrict__ z) {
  int gid = blockIdx.x * 256 + threadIdx.x;       // exactly B*H*S threads
  int bh = gid / Sc, s = gid % Sc;
  int b = bh / Hc, h = bh % Hc;
  const unsigned short* qr = qb + (size_t)(b * Sc + s) * Dc + h * DKc;
  const float* ks = ksum + bh * DKc;
  float dot = 0.f;
#pragma unroll 8
  for (int d = 0; d < DKc; ++d) dot += bf2f(qr[d]) * (ks[d] + 1e-6f);
  z[gid] = 1.f / dot;
}

// ---------------------------------------------------------------------------
// KV[bh][d][e] = sum_s K[s,d] * V[s,e]  — WMMA, M=N=64, K=S=4096 per (b,h)
// ---------------------------------------------------------------------------
__global__ __launch_bounds__(128) void attn_kv_kernel(
    const unsigned short* __restrict__ kb, const unsigned short* __restrict__ vb,
    float* __restrict__ kv) {
  int bh = blockIdx.x;
  int b = bh / Hc, h = bh % Hc;
  __shared__ unsigned short Ks[32][72];   // [s][d], 144B pitch
  __shared__ unsigned short Vs[32][72];
  const int tid  = threadIdx.x;
  const int lane = tid & 31;
  const int wave = tid >> 5;
  const int hh   = lane >> 4;
  const int m16  = lane & 15;

  CFrag c[4];
#pragma unroll
  for (int t = 0; t < 4; ++t)
#pragma unroll
    for (int q = 0; q < 8; ++q) c[t].f[q] = 0.f;

  for (int s0 = 0; s0 < Sc; s0 += 32) {
#pragma unroll
    for (int it = 0; it < 2; ++it) {
      int idx = tid * 2 + it;             // 0..255 -> 32 rows x 8 x (8 bf16)
      int r   = idx >> 3;
      int cs  = (idx & 7) * 8;
      size_t g = (size_t)(b * Sc + s0 + r) * Dc + h * DKc + cs;
      *reinterpret_cast<uint4*>(&Ks[r][cs]) = *reinterpret_cast<const uint4*>(&kb[g]);
      *reinterpret_cast<uint4*>(&Vs[r][cs]) = *reinterpret_cast<const uint4*>(&vb[g]);
    }
    __syncthreads();
    AFrag a;                              // A = K^T : A[d][s] = Ks[s][d]
#pragma unroll
    for (int i = 0; i < 16; ++i) a.u[i] = Ks[amap(i, hh)][wave * 16 + m16];
#pragma unroll
    for (int t = 0; t < 4; ++t) {
      AFrag bf;                           // B[s][e] = Vs[s][e]
#pragma unroll
      for (int i = 0; i < 16; ++i) bf.u[i] = Vs[amap(i, hh)][t * 16 + m16];
      c[t].v = __builtin_amdgcn_wmma_f32_16x16x32_bf16(
          false, a.v, false, bf.v, (short)0, c[t].v, false, false);
    }
    __syncthreads();
  }
#pragma unroll
  for (int t = 0; t < 4; ++t)
#pragma unroll
    for (int q = 0; q < 8; ++q)
      kv[((size_t)bh * DKc + wave * 16 + q + 8 * hh) * DKc + t * 16 + m16] = c[t].f[q];
}

// ---------------------------------------------------------------------------
// Vn[s,e] = z[s] * sum_d Q[s,d] * KV[d,e] — WMMA, 64 rows of s per block.
// Output written bf16 into [B,S,D] layout (input of the o-projection GEMM).
// ---------------------------------------------------------------------------
__global__ __launch_bounds__(128) void attn_out_kernel(
    const unsigned short* __restrict__ qb, const float* __restrict__ kv,
    const float* __restrict__ z, unsigned short* __restrict__ attnb) {
  int bh = blockIdx.x;
  int s0 = blockIdx.y * 64;
  int b = bh / Hc, h = bh % Hc;
  __shared__ unsigned short Qs[64][72];   // [s][d]
  __shared__ unsigned short KVs[64][72];  // [d][e]
  const int tid  = threadIdx.x;
  const int lane = tid & 31;
  const int wave = tid >> 5;
  const int hh   = lane >> 4;
  const int m16  = lane & 15;

#pragma unroll
  for (int it = 0; it < 4; ++it) {
    int idx = tid * 4 + it;               // 0..511 -> 64 rows x 8 x (8 bf16)
    int r   = idx >> 3;
    int cs  = (idx & 7) * 8;
    *reinterpret_cast<uint4*>(&Qs[r][cs]) =
        *reinterpret_cast<const uint4*>(&qb[(size_t)(b * Sc + s0 + r) * Dc + h * DKc + cs]);
  }
  const float* kvb = kv + (size_t)bh * DKc * DKc;
#pragma unroll 4
  for (int jj = 0; jj < 32; ++jj) {
    int i = tid * 32 + jj;                // 0..4095
    KVs[i >> 6][i & 63] = f2bf(kvb[i]);
  }
  __syncthreads();

  CFrag c[4];
#pragma unroll
  for (int t = 0; t < 4; ++t)
#pragma unroll
    for (int q = 0; q < 8; ++q) c[t].f[q] = 0.f;

#pragma unroll
  for (int k0 = 0; k0 < DKc; k0 += 32) {
    AFrag a;
#pragma unroll
    for (int i = 0; i < 16; ++i) a.u[i] = Qs[wave * 16 + m16][k0 + amap(i, hh)];
#pragma unroll
    for (int t = 0; t < 4; ++t) {
      AFrag bf;
#pragma unroll
      for (int i = 0; i < 16; ++i) bf.u[i] = KVs[k0 + amap(i, hh)][t * 16 + m16];
      c[t].v = __builtin_amdgcn_wmma_f32_16x16x32_bf16(
          false, a.v, false, bf.v, (short)0, c[t].v, false, false);
    }
  }
#pragma unroll
  for (int t = 0; t < 4; ++t)
#pragma unroll
    for (int q = 0; q < 8; ++q) {
      int sl = s0 + wave * 16 + q + 8 * hh;
      float zv = z[(size_t)bh * Sc + sl];
      attnb[(size_t)(b * Sc + sl) * Dc + h * DKc + t * 16 + m16] = f2bf(c[t].f[q] * zv);
    }
}

// ---------------------------------------------------------------------------
// y = LayerNorm(x + t) * g + b over D=512; one block per row.
// Writes fp32 (residual path) and optionally bf16 (next GEMM input).
// ---------------------------------------------------------------------------
__global__ __launch_bounds__(256) void add_ln_kernel(
    const float* __restrict__ x, const float* __restrict__ t,
    const float* __restrict__ g, const float* __restrict__ bvec,
    float* __restrict__ outF, unsigned short* __restrict__ outB) {
  __shared__ float s1[256], s2[256];
  size_t row = blockIdx.x;
  const float* xr = x + row * Dc;
  const float* tr = t + row * Dc;
  int tid = threadIdx.x;
  float a0 = xr[tid] + tr[tid];
  float a1 = xr[tid + 256] + tr[tid + 256];
  s1[tid] = a0 + a1;
  s2[tid] = a0 * a0 + a1 * a1;
  __syncthreads();
  for (int off = 128; off > 0; off >>= 1) {
    if (tid < off) { s1[tid] += s1[tid + off]; s2[tid] += s2[tid + off]; }
    __syncthreads();
  }
  float mean = s1[0] * (1.f / Dc);
  float var  = s2[0] * (1.f / Dc) - mean * mean;
  float rs   = rsqrtf(var + 1e-5f);
  float o0 = (a0 - mean) * rs * g[tid] + bvec[tid];
  float o1 = (a1 - mean) * rs * g[tid + 256] + bvec[tid + 256];
  outF[row * Dc + tid] = o0;
  outF[row * Dc + tid + 256] = o1;
  if (outB) {
    outB[row * Dc + tid] = f2bf(o0);
    outB[row * Dc + tid + 256] = f2bf(o1);
  }
}

// ---------------------------------------------------------------------------
extern "C" void kernel_launch(void* const* d_in, const int* in_sizes, int n_in,
                              void* d_out, int out_size, void* d_ws, size_t ws_size,
                              hipStream_t stream) {
  (void)in_sizes; (void)n_in; (void)out_size; (void)ws_size;
  const float* x    = (const float*)d_in[0];
  const float* wq   = (const float*)d_in[1];
  const float* bq   = (const float*)d_in[2];
  const float* wk   = (const float*)d_in[3];
  const float* bk   = (const float*)d_in[4];
  const float* wv   = (const float*)d_in[5];
  const float* bvv  = (const float*)d_in[6];
  const float* wo   = (const float*)d_in[7];
  const float* bo   = (const float*)d_in[8];
  const float* qcw  = (const float*)d_in[9];
  const float* qcb  = (const float*)d_in[10];
  const float* kcw  = (const float*)d_in[11];
  const float* kcb  = (const float*)d_in[12];
  const float* vcw  = (const float*)d_in[13];
  const float* vcb  = (const float*)d_in[14];
  const float* w1   = (const float*)d_in[15];
  const float* b1   = (const float*)d_in[16];
  const float* w2   = (const float*)d_in[17];
  const float* b2   = (const float*)d_in[18];
  const float* ln1g = (const float*)d_in[19];
  const float* ln1b = (const float*)d_in[20];
  const float* ln2g = (const float*)d_in[21];
  const float* ln2b = (const float*)d_in[22];

  char* wsp = (char*)d_ws;
  auto carve = [&](size_t bytes) -> char* {
    char* p = wsp;
    wsp += (bytes + 255) & ~size_t(255);
    return p;
  };
  const size_t tokD  = (size_t)Mtot * Dc;            // 16.78M
  unsigned short* xb   = (unsigned short*)carve(tokD * 2);
  unsigned short* wqb  = (unsigned short*)carve((size_t)Dc * Dc * 2);
  unsigned short* wkb  = (unsigned short*)carve((size_t)Dc * Dc * 2);
  unsigned short* wvb  = (unsigned short*)carve((size_t)Dc * Dc * 2);
  unsigned short* wob  = (unsigned short*)carve((size_t)Dc * Dc * 2);
  unsigned short* w1b  = (unsigned short*)carve((size_t)DFFc * Dc * 2);
  unsigned short* w2b  = (unsigned short*)carve((size_t)Dc * DFFc * 2);
  float* qraw = (float*)carve(tokD * 4);             // later reused: o-proj out
  float* kraw = (float*)carve(tokD * 4);             // later reused: FFN2 out
  float* vraw = (float*)carve(tokD * 4);             // later reused: LN1 f32 out
  unsigned short* qb = (unsigned short*)carve(tokD * 2);
  unsigned short* kb = (unsigned short*)carve(tokD * 2);
  unsigned short* vb = (unsigned short*)carve(tokD * 2);
  float* kvbuf = (float*)carve((size_t)Bc * Hc * DKc * DKc * 4);
  float* ksum  = (float*)carve((size_t)Bc * Hc * DKc * 4);
  float* zbuf  = (float*)carve((size_t)Bc * Hc * Sc * 4);
  unsigned short* attnb = (unsigned short*)carve(tokD * 2);
  unsigned short* x1b   = (unsigned short*)carve(tokD * 2);
  unsigned short* hb    = (unsigned short*)carve((size_t)Mtot * DFFc * 2);
  float* oattn = qraw;
  float* hff   = kraw;
  float* x1f   = vraw;

  auto cvt = [&](const float* in, unsigned short* out, int n) {
    cvt_kernel<<<dim3((n + 255) / 256), dim3(256), 0, stream>>>(in, out, n);
  };
  auto gemm = [&](const unsigned short* Am, const unsigned short* Wm, const float* bb,
                  float* of, unsigned short* ob, int K, int N, int mode) {
    gemm_bf16_wmma<<<dim3(N / 128, Mtot / 128), dim3(256), 0, stream>>>(
        Am, Wm, bb, of, ob, K, N, mode);
  };

  // 1) precision downcast of activations + weights
  cvt(x, xb, Mtot * Dc);
  cvt(wq, wqb, Dc * Dc);
  cvt(wk, wkb, Dc * Dc);
  cvt(wv, wvb, Dc * Dc);
  cvt(wo, wob, Dc * Dc);
  cvt(w1, w1b, DFFc * Dc);
  cvt(w2, w2b, Dc * DFFc);

  // 2) q/k/v projections (WMMA, async double-buffered)
  gemm(xb, wqb, bq,  qraw, nullptr, Dc, Dc, 0);
  gemm(xb, wkb, bk,  kraw, nullptr, Dc, Dc, 0);
  gemm(xb, wvb, bvv, vraw, nullptr, Dc, Dc, 0);

  // 3) depthwise causal conv (+ elu+1 on q/k) -> bf16
  int nconv = Mtot * Dc;
  dwconv_kernel<<<dim3(nconv / 256), dim3(256), 0, stream>>>(qraw, qcw, qcb, qb, 1);
  dwconv_kernel<<<dim3(nconv / 256), dim3(256), 0, stream>>>(kraw, kcw, kcb, kb, 1);
  dwconv_kernel<<<dim3(nconv / 256), dim3(256), 0, stream>>>(vraw, vcw, vcb, vb, 0);

  // 4) ksum and Z normalizer
  ksum_kernel<<<dim3(Bc * Hc), dim3(256), 0, stream>>>(kb, ksum);
  z_kernel<<<dim3(Bc * Hc * Sc / 256), dim3(256), 0, stream>>>(qb, ksum, zbuf);

  // 5) KV = K^T V per head (WMMA), then Vn = z * (Q . KV) (WMMA)
  attn_kv_kernel<<<dim3(Bc * Hc), dim3(128), 0, stream>>>(kb, vb, kvbuf);
  attn_out_kernel<<<dim3(Bc * Hc, Sc / 64), dim3(128), 0, stream>>>(qb, kvbuf, zbuf, attnb);

  // 6) output projection + residual + LN1
  gemm(attnb, wob, bo, oattn, nullptr, Dc, Dc, 0);
  add_ln_kernel<<<dim3(Mtot), dim3(256), 0, stream>>>(x, oattn, ln1g, ln1b, x1f, x1b);

  // 7) conv-FFN: W1 + GELU^2 (WMMA, bf16 epilogue), W2 (WMMA), residual + LN2
  gemm(x1b, w1b, b1, nullptr, hb, Dc, DFFc, 1);
  gemm(hb, w2b, b2, hff, nullptr, DFFc, Dc, 0);
  add_ln_kernel<<<dim3(Mtot), dim3(256), 0, stream>>>(x1f, hff, ln2g, ln2b,
                                                      (float*)d_out, nullptr);
}
